// MambaLayer_2405181685823
// MI455X (gfx1250) — matrix operand order repaired and checked by reading, not compiled, not run on hardware
//
#include <hip/hip_runtime.h>
#include <stdint.h>

// ---------------- problem constants ----------------
#define BSZ   2
#define DIMC  192          // channels
#define LSEQ  16384        // 16*32*32
#define DIN   384          // d_inner
#define E2    768          // 2*d_inner
#define DTR   12           // dt_rank
#define DST   16           // d_state
#define XDB   48           // padded x_dbl row (44 -> 48)
#define NCH   64           // scan chunks
#define LCH   (LSEQ/NCH)   // 256 steps per chunk
#define MROWS (BSZ*LSEQ)   // 32768 GEMM rows
#define NSL   16
#define LS    (LSEQ/NSL)   // 1024

typedef __bf16 bhalf;
typedef __attribute__((ext_vector_type(16))) __bf16 v16bf;
typedef __attribute__((ext_vector_type(8)))  __bf16 v8bf;
typedef __attribute__((ext_vector_type(8)))  float  v8f;

static __device__ __forceinline__ bhalf f2bf(float f) {
  uint32_t u = __builtin_bit_cast(uint32_t, f);
  uint32_t r = u + 0x7FFFu + ((u >> 16) & 1u);      // round-to-nearest-even
  uint16_t h = (uint16_t)(r >> 16);
  return __builtin_bit_cast(bhalf, h);
}
static __device__ __forceinline__ float bf2f(bhalf b) {
  uint16_t h = __builtin_bit_cast(uint16_t, b);
  uint32_t u = ((uint32_t)h) << 16;
  return __builtin_bit_cast(float, u);
}
static __device__ __forceinline__ float silu_f(float v) { return v / (1.0f + __expf(-v)); }

// scan-time t -> original sequence index l, per branch (fwd / reversed / slice-transposed)
static __device__ __forceinline__ int mapidx(int br, int t) {
  if (br == 0) return t;
  if (br == 1) return LSEQ - 1 - t;
  return (t & (NSL - 1)) * LS + (t >> 4);
}

// ---------------- WMMA bf16 fragments (CDNA5 16x16x32, wave32) ----------------
// A (16x32): lane m=lane&15, hi=lane>>4 holds k = hi*8+j (j<8) and 16+hi*8+(j-8)
static __device__ __forceinline__ v16bf load_a_frag(const bhalf* A, int lda, int lane) {
  int m = lane & 15, hi = lane >> 4;
  const bhalf* p = A + (size_t)m * lda + hi * 8;
  union { v16bf v; v8bf h[2]; } u;
  u.h[0] = *(const v8bf*)p;
  u.h[1] = *(const v8bf*)(p + 16);
  return u.v;
}
// B (32x16) from row-major weight W[n][k]: lane n=lane&15, hi=lane>>4 holds k = hi*8+j (j<8) and 16+hi*8+(j-8)
static __device__ __forceinline__ v16bf load_b_frag(const bhalf* W, int ldw, int lane) {
  int n = lane & 15, hi = lane >> 4;
  const bhalf* p = W + (size_t)n * ldw + hi * 8;
  union { v16bf v; v8bf h[2]; } u;
  u.h[0] = *(const v8bf*)p;
  u.h[1] = *(const v8bf*)(p + 16);
  return u.v;
}
static __device__ __forceinline__ v8f wmma_bf16(v16bf a, v16bf b, v8f c) {
  return __builtin_amdgcn_wmma_f32_16x16x32_bf16(false, a, false, b, (short)0, c, false, false);
}

// ---------------- kernels ----------------

// weight prep: bf16 conversions + A = -exp(A_log)
__global__ void k_prep(const float* in_w, const float* out_w, const float* xproj_w,
                       const float* A_log,
                       bhalf* wz, bhalf* wout, bhalf* wproj, float* Aneg) {
  int i = blockIdx.x * blockDim.x + threadIdx.x;
  const int n1 = E2 * DIMC, n2 = DIMC * DIN, n3 = 3 * XDB * DIN, n4 = 3 * DIN * DST;
  if (i < n1) { wz[i] = f2bf(in_w[i]); return; }
  i -= n1;
  if (i < n2) { wout[i] = f2bf(out_w[i]); return; }
  i -= n2;
  if (i < n3) {
    int br = i / (XDB * DIN); int r = i % (XDB * DIN); int e = r / DIN; int d = r % DIN;
    float v = (e < DTR + 2 * DST) ? xproj_w[((size_t)br * (DTR + 2 * DST) + e) * DIN + d] : 0.0f;
    wproj[i] = f2bf(v); return;
  }
  i -= n3;
  if (i < n4) Aneg[i] = -__expf(A_log[i]);
}

// channel layernorm: x (B,C,L) -> xn rows (B*L, 192) bf16
__global__ void k_ln(const float* x, const float* gamma, const float* beta, bhalf* xn) {
  int idx = blockIdx.x * blockDim.x + threadIdx.x;
  if (idx >= MROWS) return;
  int b = idx / LSEQ, l = idx % LSEQ;
  const float* xp = x + (size_t)b * DIMC * LSEQ + l;
  float s = 0.f, q = 0.f;
  for (int c = 0; c < DIMC; ++c) { float v = xp[(size_t)c * LSEQ]; s += v; q += v * v; }
  float mu = s * (1.0f / DIMC);
  float var = q * (1.0f / DIMC) - mu * mu;
  float rs = rsqrtf(var + 1e-5f);
  bhalf* o = xn + (size_t)idx * DIMC;
  for (int c = 0; c < DIMC; ++c)
    o[c] = f2bf((xp[(size_t)c * LSEQ] - mu) * rs * gamma[c] + beta[c]);
}

// in_proj: per-wave 16x64 tile, K=192.
//   e < 384  (xh half): store (b,e,l) layout for the conv's sequence gather
//   e >= 384 (z half) : store rows (b*L, d) layout for coalesced gating reads
__global__ void k_gemm_xz(const bhalf* __restrict__ xn, const bhalf* __restrict__ wz,
                          bhalf* __restrict__ xzh, bhalf* __restrict__ zrows) {
  int lane = threadIdx.x & 31, wave = threadIdx.x >> 5;
  int m0 = blockIdx.x * 128 + wave * 16;
  int n0 = blockIdx.y * 64;
  v8f acc[4] = {};
  for (int k = 0; k < DIMC; k += 32) {
    v16bf a = load_a_frag(xn + (size_t)m0 * DIMC + k, DIMC, lane);
#pragma unroll
    for (int j = 0; j < 4; ++j) {
      v16bf bm = load_b_frag(wz + (size_t)(n0 + j * 16) * DIMC + k, DIMC, lane);
      acc[j] = wmma_bf16(a, bm, acc[j]);
    }
  }
  int hi = lane >> 4, n = lane & 15;
  int mrow = m0 + hi * 8;
  if (n0 < DIN) {
    int b = mrow / LSEQ, l0 = mrow % LSEQ;
#pragma unroll
    for (int j = 0; j < 4; ++j) {
      int e = n0 + j * 16 + n;
      v8bf o;
#pragma unroll
      for (int r = 0; r < 8; ++r) o[r] = f2bf(acc[j][r]);  // rows hi*8+r = consecutive l
      *(v8bf*)(xzh + ((size_t)b * DIN + e) * LSEQ + l0) = o;
    }
  } else {
#pragma unroll
    for (int j = 0; j < 4; ++j) {
      int d = n0 - DIN + j * 16 + n;
#pragma unroll
      for (int r = 0; r < 8; ++r)
        zrows[(size_t)(mrow + r) * DIN + d] = f2bf(acc[j][r]);
    }
  }
}

// depthwise causal conv(k=4)+SiLU with branch gather; writes xc rows (B*L, 384) bf16
__global__ void k_conv(int br, const bhalf* xzh, const float* conv_w, const float* conv_b,
                       bhalf* xc) {
  int idx = blockIdx.x * blockDim.x + threadIdx.x;
  if (idx >= BSZ * LSEQ * DIN) return;
  int d = idx % DIN;
  int t = (idx / DIN) % LSEQ;
  int b = idx / (DIN * LSEQ);
  const bhalf* xrow = xzh + ((size_t)b * DIN + d) * LSEQ;
  float s = conv_b[br * DIN + d];
  const float* w = conv_w + ((size_t)br * DIN + d) * 4;
#pragma unroll
  for (int k = 0; k < 4; ++k) {
    int u = t - 3 + k;
    if (u >= 0) s += w[k] * bf2f(xrow[mapidx(br, u)]);
  }
  xc[((size_t)b * LSEQ + t) * DIN + d] = f2bf(silu_f(s));
}

// x_proj: xdbl rows (B*L, 48) f32 = xc rows (K=384) x wproj (48x384)
__global__ void k_gemm_xdbl(const bhalf* __restrict__ xc, const bhalf* __restrict__ wproj,
                            float* __restrict__ xdbl) {
  int lane = threadIdx.x & 31, wave = threadIdx.x >> 5;
  int m0 = blockIdx.x * 128 + wave * 16;
  v8f acc[3] = {};
  for (int k = 0; k < DIN; k += 32) {
    v16bf a = load_a_frag(xc + (size_t)m0 * DIN + k, DIN, lane);
#pragma unroll
    for (int j = 0; j < 3; ++j) {
      v16bf bm = load_b_frag(wproj + (size_t)(j * 16) * DIN + k, DIN, lane);
      acc[j] = wmma_bf16(a, bm, acc[j]);
    }
  }
  int hi = lane >> 4, n = lane & 15;
#pragma unroll
  for (int j = 0; j < 3; ++j)
#pragma unroll
    for (int r = 0; r < 8; ++r)
      xdbl[(size_t)(m0 + hi * 8 + r) * XDB + j * 16 + n] = acc[j][r];
}

// delta = softplus(dt . dt_w + dt_b), layout (b, t, d) -- writer and scan reader coalesced
__global__ void k_delta(int br, const float* xdbl, const float* dt_w, const float* dt_b,
                        float* delta) {
  int idx = blockIdx.x * blockDim.x + threadIdx.x;
  if (idx >= BSZ * LSEQ * DIN) return;
  int d = idx % DIN;
  int t = (idx / DIN) % LSEQ;
  int b = idx / (DIN * LSEQ);
  const float* row = xdbl + (size_t)(b * LSEQ + t) * XDB;   // broadcast across lanes
  const float* w = dt_w + ((size_t)br * DIN + d) * DTR;
  float s = dt_b[br * DIN + d];
#pragma unroll
  for (int r = 0; r < DTR; ++r) s += row[r] * w[r];
  delta[((size_t)b * LSEQ + t) * DIN + d] = (s > 20.f) ? s : log1pf(__expf(s));
}

// chunked linear scan, pass A: local scan with h0=0 + chunk decay product.
// Threads are d-fast: all 32 lanes of a wave share t each step, so the xdbl
// B-row is a uniform broadcast and delta/xc reads are coalesced.
__global__ void k_scanA(int br, const float* delta, const bhalf* xc, const float* xdbl,
                        const float* Aneg, float* hbuf, float* Pbuf) {
  int idx = blockIdx.x * blockDim.x + threadIdx.x;
  if (idx >= BSZ * NCH * DIN) return;
  int d = idx % DIN;
  int c = (idx / DIN) % NCH;
  int b = idx / (DIN * NCH);
  float Av[DST], h[DST], P[DST];
  const float* Ap = Aneg + ((size_t)br * DIN + d) * DST;
#pragma unroll
  for (int n = 0; n < DST; ++n) { Av[n] = Ap[n]; h[n] = 0.f; P[n] = 1.f; }
  for (int t = c * LCH; t < (c + 1) * LCH; ++t) {
    size_t rowi = (size_t)b * LSEQ + t;
    float dt = delta[rowi * DIN + d];
    float xv = bf2f(xc[rowi * DIN + d]);
    const float* row = xdbl + rowi * XDB;
    float du = dt * xv;
#pragma unroll
    for (int n = 0; n < DST; ++n) {
      float a = __expf(dt * Av[n]);
      h[n] = a * h[n] + du * row[DTR + n];
      P[n] *= a;
    }
  }
  float* ho = hbuf + (size_t)idx * DST;
  float* Po = Pbuf + (size_t)idx * DST;
#pragma unroll
  for (int n = 0; n < DST; ++n) { ho[n] = h[n]; Po[n] = P[n]; }
}

// pass B: sequential combine over chunks -> carried-in state per chunk
__global__ void k_comb(const float* hbuf, const float* Pbuf, float* Hin) {
  int idx = blockIdx.x * blockDim.x + threadIdx.x;
  if (idx >= BSZ * DIN * DST) return;
  int n = idx % DST;
  int d = (idx / DST) % DIN;
  int b = idx / (DST * DIN);
  float H = 0.f;
  for (int c = 0; c < NCH; ++c) {
    size_t base = (((size_t)(b * NCH + c) * DIN) + d) * DST + n;
    Hin[base] = H;
    H = Pbuf[base] * H + hbuf[base];
  }
}

// pass C: rescan with correct h0, project with C, add D skip, gate with silu(z),
// scatter into row-major f32 accumulator (all lanes share l -> coalesced)
__global__ void k_scanC(int br, int accumulate, const float* delta, const bhalf* xc,
                        const float* xdbl, const float* Aneg, const float* Hin,
                        const bhalf* zrows, const float* Dp, float* accrows) {
  int idx = blockIdx.x * blockDim.x + threadIdx.x;
  if (idx >= BSZ * NCH * DIN) return;
  int d = idx % DIN;
  int c = (idx / DIN) % NCH;
  int b = idx / (DIN * NCH);
  float Av[DST], h[DST];
  const float* Ap = Aneg + ((size_t)br * DIN + d) * DST;
  const float* Hp = Hin + (size_t)idx * DST;
#pragma unroll
  for (int n = 0; n < DST; ++n) { Av[n] = Ap[n]; h[n] = Hp[n]; }
  float Dv = Dp[br * DIN + d];
  for (int t = c * LCH; t < (c + 1) * LCH; ++t) {
    size_t rowi = (size_t)b * LSEQ + t;
    float dt = delta[rowi * DIN + d];
    float xv = bf2f(xc[rowi * DIN + d]);
    const float* row = xdbl + rowi * XDB;
    float du = dt * xv;
    float y = 0.f;
#pragma unroll
    for (int n = 0; n < DST; ++n) {
      float a = __expf(dt * Av[n]);
      h[n] = a * h[n] + du * row[DTR + n];
      y += h[n] * row[DTR + DST + n];
    }
    float outv = y + Dv * xv;
    int l = mapidx(br, t);
    size_t orow = (size_t)b * LSEQ + l;
    float val = outv * silu_f(bf2f(zrows[orow * DIN + d]));
    size_t oi = orow * DIN + d;
    if (accumulate) accrows[oi] += val; else accrows[oi] = val;
  }
}

// row-major f32 accumulator -> bf16 rows for out_proj (fully coalesced)
__global__ void k_cvt(const float* accrows, bhalf* accr) {
  int idx = blockIdx.x * blockDim.x + threadIdx.x;
  if (idx >= MROWS * DIN) return;
  accr[idx] = f2bf(accrows[idx]);
}

// out_proj + residual: out[b,c,l] = sum_d accr[m][d]*out_w[c][d] + x[b,c,l]
__global__ void k_gemm_out(const bhalf* __restrict__ accr, const bhalf* __restrict__ wout,
                           const float* __restrict__ x, float* __restrict__ out) {
  int lane = threadIdx.x & 31, wave = threadIdx.x >> 5;
  int m0 = blockIdx.x * 128 + wave * 16;
  int n0 = blockIdx.y * 96;
  v8f acc[6] = {};
  for (int k = 0; k < DIN; k += 32) {
    v16bf a = load_a_frag(accr + (size_t)m0 * DIN + k, DIN, lane);
#pragma unroll
    for (int j = 0; j < 6; ++j) {
      v16bf bm = load_b_frag(wout + (size_t)(n0 + j * 16) * DIN + k, DIN, lane);
      acc[j] = wmma_bf16(a, bm, acc[j]);
    }
  }
  int hi = lane >> 4, n = lane & 15;
  int mrow = m0 + hi * 8;
  int b = mrow / LSEQ, l0 = mrow % LSEQ;
#pragma unroll
  for (int j = 0; j < 6; ++j) {
    int cch = n0 + j * 16 + n;
    size_t base = ((size_t)b * DIMC + cch) * LSEQ + l0;
#pragma unroll
    for (int r = 0; r < 8; ++r)
      out[base + r] = acc[j][r] + x[base + r];
  }
}

// ---------------- host ----------------
extern "C" void kernel_launch(void* const* d_in, const int* in_sizes, int n_in,
                              void* d_out, int out_size, void* d_ws, size_t ws_size,
                              hipStream_t stream) {
  const float* x      = (const float*)d_in[0];
  const float* gamma  = (const float*)d_in[1];
  const float* beta   = (const float*)d_in[2];
  const float* in_w   = (const float*)d_in[3];
  const float* out_w  = (const float*)d_in[4];
  const float* conv_w = (const float*)d_in[5];
  const float* conv_b = (const float*)d_in[6];
  const float* xproj  = (const float*)d_in[7];
  const float* dt_w   = (const float*)d_in[8];
  const float* dt_b   = (const float*)d_in[9];
  const float* A_log  = (const float*)d_in[10];
  const float* Dp     = (const float*)d_in[11];
  float* out = (float*)d_out;
  (void)in_sizes; (void)n_in; (void)out_size;

  // workspace layout (~240 MB total), 256B aligned slices
  char* w = (char*)d_ws;
  size_t off = 0;
  auto alloc = [&](size_t bytes) -> char* {
    char* p = w + off;
    off = (off + bytes + 255) & ~(size_t)255;
    return p;
  };
  bhalf* xn      = (bhalf*)alloc((size_t)MROWS * DIMC * 2);
  bhalf* wz      = (bhalf*)alloc((size_t)E2 * DIMC * 2);
  bhalf* woutb   = (bhalf*)alloc((size_t)DIMC * DIN * 2);
  bhalf* wproj   = (bhalf*)alloc((size_t)3 * XDB * DIN * 2);
  float* Aneg    = (float*)alloc((size_t)3 * DIN * DST * 4);
  bhalf* xzh     = (bhalf*)alloc((size_t)BSZ * DIN * LSEQ * 2);   // xh half, (b,e,l)
  bhalf* zrows   = (bhalf*)alloc((size_t)MROWS * DIN * 2);        // z half, rows
  bhalf* xc      = (bhalf*)alloc((size_t)MROWS * DIN * 2);
  float* xdbl    = (float*)alloc((size_t)MROWS * XDB * 4);
  float* delta   = (float*)alloc((size_t)MROWS * DIN * 4);
  float* hbuf    = (float*)alloc((size_t)BSZ * DIN * NCH * DST * 4);
  float* Pbuf    = (float*)alloc((size_t)BSZ * DIN * NCH * DST * 4);
  float* Hin     = (float*)alloc((size_t)BSZ * DIN * NCH * DST * 4);
  float* accrows = (float*)alloc((size_t)MROWS * DIN * 4);
  bhalf* accr    = (bhalf*)alloc((size_t)MROWS * DIN * 2);
  (void)ws_size;

  const int prep_tot = E2 * DIMC + DIMC * DIN + 3 * XDB * DIN + 3 * DIN * DST;
  k_prep<<<(prep_tot + 255) / 256, 256, 0, stream>>>(in_w, out_w, xproj, A_log,
                                                     wz, woutb, wproj, Aneg);
  k_ln<<<(MROWS + 255) / 256, 256, 0, stream>>>(x, gamma, beta, xn);
  k_gemm_xz<<<dim3(MROWS / 128, E2 / 64), 256, 0, stream>>>(xn, wz, xzh, zrows);

  const int totel = BSZ * LSEQ * DIN;
  const int nsc = BSZ * NCH * DIN;
  const int ncb = BSZ * DIN * DST;
  for (int br = 0; br < 3; ++br) {
    k_conv<<<(totel + 255) / 256, 256, 0, stream>>>(br, xzh, conv_w, conv_b, xc);
    k_gemm_xdbl<<<MROWS / 128, 256, 0, stream>>>(xc, wproj + (size_t)br * XDB * DIN, xdbl);
    k_delta<<<(totel + 255) / 256, 256, 0, stream>>>(br, xdbl, dt_w, dt_b, delta);
    k_scanA<<<(nsc + 255) / 256, 256, 0, stream>>>(br, delta, xc, xdbl, Aneg, hbuf, Pbuf);
    k_comb<<<(ncb + 255) / 256, 256, 0, stream>>>(hbuf, Pbuf, Hin);
    k_scanC<<<(nsc + 255) / 256, 256, 0, stream>>>(br, br > 0 ? 1 : 0, delta, xc, xdbl,
                                                   Aneg, Hin, zrows, Dp, accrows);
  }
  k_cvt<<<(totel + 255) / 256, 256, 0, stream>>>(accrows, accr);
  k_gemm_out<<<dim3(MROWS / 128, 2), 256, 0, stream>>>(accr, woutb, x, out);
}
